// Dense_test_4layer_40510131536142
// MI455X (gfx1250) — matrix operand order repaired and checked by reading, not compiled, not run on hardware
//
#include <hip/hip_runtime.h>

typedef __attribute__((ext_vector_type(16))) _Float16 v16h;
typedef __attribute__((ext_vector_type(8)))  float    v8f;

#define TT      100
#define BB      128
#define IN_DIM  2752
#define NH1     512
#define NH2     512
#define NH3     256
#define NCLS    100
#define NCLSP   128   // padded to tile multiple

// ---------------- prep kernels ----------------

__global__ void k_prep_weight(const float* __restrict__ W, const float* __restrict__ mask,
                              _Float16* __restrict__ out, int rows_src, int rows_dst, int cols) {
  int idx = blockIdx.x * blockDim.x + threadIdx.x;
  int total = rows_dst * cols;
  if (idx >= total) return;
  int r = idx / cols;
  float v = 0.f;
  if (r < rows_src) {
    v = W[idx];
    if (mask) v *= mask[idx];
  }
  out[idx] = (_Float16)v;
}

// dvs (B,T,2752) -> X (T*B, 2752) f16
__global__ void k_prep_x(const float* __restrict__ dvs, _Float16* __restrict__ X) {
  size_t idx = (size_t)blockIdx.x * blockDim.x + threadIdx.x;
  size_t total = (size_t)TT * BB * IN_DIM;
  if (idx >= total) return;
  int f = (int)(idx % IN_DIM);
  size_t tb = idx / IN_DIM;
  int b = (int)(tb % BB);
  int t = (int)(tb / BB);
  X[idx] = (_Float16)dvs[(size_t)b * TT * IN_DIM + (size_t)t * IN_DIM + f];
}

__global__ void k_sigmoid(const float* __restrict__ in, float* __restrict__ out, int n) {
  int i = blockIdx.x * blockDim.x + threadIdx.x;
  if (i < n) out[i] = 1.0f / (1.0f + __expf(-in[i]));
}

__global__ void k_pad_bias(const float* __restrict__ b, float* __restrict__ out, int n, int np) {
  int i = blockIdx.x * blockDim.x + threadIdx.x;
  if (i < np) out[i] = (i < n) ? b[i] : 0.f;
}

__global__ void k_copy_f32(const float* __restrict__ src, float* __restrict__ dst, int n) {
  int i = blockIdx.x * blockDim.x + threadIdx.x;
  if (i < n) dst[i] = src[i];
}

__global__ void k_zero_f32(float* __restrict__ p, int n) {
  int i = blockIdx.x * blockDim.x + threadIdx.x;
  if (i < n) p[i] = 0.f;
}

__global__ void k_zero_f16(_Float16* __restrict__ p, int n) {
  int i = blockIdx.x * blockDim.x + threadIdx.x;
  if (i < n) p[i] = (_Float16)0.f;
}

// ---------------- shared WMMA helpers ----------------
// Fragment addressing per CDNA5 ISA VGPR layouts (CDNA5 ISA):
// A 16x32 f16: lane(0..15) row M=lane&15, halves K = kb..kb+7 then kb+16..kb+23, kb=(lane>>4)*8
// B 32x16 f16 (from row-major N x K weights): lane col N=lane&15, halves K = kb..kb+7 then kb+16..kb+23, kb=(lane>>4)*8

__device__ __forceinline__ void load_a_frag(v16h& a, const _Float16* p) {
  uint4* d = (uint4*)&a;
  d[0] = *(const uint4*)p;         // K = kb .. kb+7
  d[1] = *(const uint4*)(p + 16);  // K = kb+16 .. kb+23
}

__device__ __forceinline__ void load_b_frag(v16h& b, const _Float16* p) {
  uint4* d = (uint4*)&b;
  d[0] = *(const uint4*)p;         // K = kb .. kb+7
  d[1] = *(const uint4*)(p + 16);  // K = kb+16 .. kb+23
}

// ---------------- direct-load WMMA GEMM (small per-step GEMMs; operands L2-resident) -------
// C(MxN,f32) = A(MxK,f16,row) @ W(NxK,f16,row)^T + bias[n] (+C)
// block = 256 threads = 8 waves; block tile 128(M) x 128(N); wave tile 32 x 64.
// M % 128 == 0, N % 128 == 0, K % 32 == 0 (all shapes padded by construction).

__global__ __launch_bounds__(256)
void k_gemm_wmma(const _Float16* __restrict__ A, const _Float16* __restrict__ W,
                 const float* __restrict__ bias, float* __restrict__ C,
                 int M, int N, int K, int accumulate) {
  const int lane = threadIdx.x & 31;
  const int wave = threadIdx.x >> 5;
  const int wm = wave & 3;   // 4 waves along M
  const int wn = wave >> 2;  // 2 waves along N
  const int m0 = blockIdx.x * 128 + wm * 32;
  const int n0 = blockIdx.y * 128 + wn * 64;

  v8f acc[2][4];
#pragma unroll
  for (int i = 0; i < 2; ++i)
#pragma unroll
    for (int j = 0; j < 4; ++j)
#pragma unroll
      for (int r = 0; r < 8; ++r) acc[i][j][r] = 0.f;

  const int arow = lane & 15;
  const int akb  = (lane >> 4) << 3;
  const int brow = lane & 15;
  const int bkb  = (lane >> 4) << 3;

  for (int k0 = 0; k0 < K; k0 += 32) {
    v16h a[2], b[4];
#pragma unroll
    for (int i = 0; i < 2; ++i)
      load_a_frag(a[i], A + (size_t)(m0 + i * 16 + arow) * K + (k0 + akb));
#pragma unroll
    for (int j = 0; j < 4; ++j) {
      const _Float16* p = W + (size_t)(n0 + j * 16 + brow) * K + (k0 + bkb);
      load_b_frag(b[j], p);
      __builtin_prefetch(p + 32, 0, 1);  // next K tile of the weight stream
    }
#pragma unroll
    for (int i = 0; i < 2; ++i)
#pragma unroll
      for (int j = 0; j < 4; ++j)
        acc[i][j] = __builtin_amdgcn_wmma_f32_16x16x32_f16(
            false, a[i], false, b[j], (short)0, acc[i][j], false, false);
  }

  const int crow = (lane >> 4) << 3;
  const int ccol = lane & 15;
#pragma unroll
  for (int i = 0; i < 2; ++i) {
#pragma unroll
    for (int j = 0; j < 4; ++j) {
#pragma unroll
      for (int r = 0; r < 8; ++r) {
        size_t mm = (size_t)(m0 + i * 16 + crow + r);
        int    nn = n0 + j * 16 + ccol;
        float v = acc[i][j][r] + bias[nn];
        if (accumulate) v += C[mm * N + nn];
        C[mm * N + nn] = v;
      }
    }
  }
}

// ---------------- LDS-staged WMMA GEMM with async global->LDS copies (big L1 GEMM) ----------
// Double-buffered 128x32 A/B tiles staged via GLOBAL_LOAD_ASYNC_TO_LDS_B128 (ASYNCcnt),
// padded LDS rows (stride 40 halfs) for conflict-free b128 fragment reads.

#define LSTR 40  // halfs per LDS tile row (32 data + 8 pad), keeps rows 16B aligned

// One tile = 128 rows x 32 halfs = 512 x 16B chunks; 8 waves x 2 instr x 32 lanes.
__device__ __forceinline__ void async_copy_tile(const _Float16* __restrict__ g, int ldg,
                                                int row_base, int k0,
                                                _Float16* lds_tile, int tid) {
#pragma unroll
  for (int i = 0; i < 2; ++i) {
    int c   = tid + 256 * i;       // chunk id 0..511
    int row = c >> 2;              // 4 chunks per row
    int col = (c & 3) << 3;        // half offset within row
    const _Float16* gp = g + (size_t)(row_base + row) * ldg + (k0 + col);
    unsigned lds_addr = (unsigned)(unsigned long long)(uintptr_t)(lds_tile + row * LSTR + col);
    unsigned long long ga = (unsigned long long)(uintptr_t)gp;
    asm volatile("global_load_async_to_lds_b128 %0, %1, off"
                 :: "v"(lds_addr), "v"(ga)
                 : "memory");
  }
}

__global__ __launch_bounds__(256)
void k_gemm_wmma_lds(const _Float16* __restrict__ A, const _Float16* __restrict__ W,
                     const float* __restrict__ bias, float* __restrict__ C,
                     int M, int N, int K) {
  __shared__ _Float16 sA[2][128 * LSTR];
  __shared__ _Float16 sB[2][128 * LSTR];

  const int tid  = threadIdx.x;
  const int lane = tid & 31;
  const int wave = tid >> 5;
  const int wm = wave & 3;   // 4 waves along M
  const int wn = wave >> 2;  // 2 waves along N
  const int mBlk = blockIdx.x * 128;
  const int nBlk = blockIdx.y * 128;

  v8f acc[2][4];
#pragma unroll
  for (int i = 0; i < 2; ++i)
#pragma unroll
    for (int j = 0; j < 4; ++j)
#pragma unroll
      for (int r = 0; r < 8; ++r) acc[i][j][r] = 0.f;

  const int arow = lane & 15;
  const int akb  = (lane >> 4) << 3;
  const int brow = lane & 15;
  const int bkb  = (lane >> 4) << 3;

  // prologue: stage tile 0 (4 async instructions per wave)
  async_copy_tile(A, K, mBlk, 0, &sA[0][0], tid);
  async_copy_tile(W, K, nBlk, 0, &sB[0][0], tid);

  int buf = 0;
  for (int k0 = 0; k0 < K; k0 += 32) {
    if (k0 + 32 < K) {
      // overlap: issue next tile into the other buffer, then wait only for current tile.
      async_copy_tile(A, K, mBlk, k0 + 32, &sA[buf ^ 1][0], tid);
      async_copy_tile(W, K, nBlk, k0 + 32, &sB[buf ^ 1][0], tid);
      // async loads complete in order: <=4 outstanding => current tile's 4 are done.
      asm volatile("s_wait_asynccnt 0x4" ::: "memory");
    } else {
      asm volatile("s_wait_asynccnt 0x0" ::: "memory");
    }
    __syncthreads();

    v16h a[2], b[4];
#pragma unroll
    for (int i = 0; i < 2; ++i)
      load_a_frag(a[i], &sA[buf][(wm * 32 + i * 16 + arow) * LSTR + akb]);
#pragma unroll
    for (int j = 0; j < 4; ++j)
      load_b_frag(b[j], &sB[buf][(wn * 64 + j * 16 + brow) * LSTR + bkb]);

#pragma unroll
    for (int i = 0; i < 2; ++i)
#pragma unroll
      for (int j = 0; j < 4; ++j)
        acc[i][j] = __builtin_amdgcn_wmma_f32_16x16x32_f16(
            false, a[i], false, b[j], (short)0, acc[i][j], false, false);

    __syncthreads();  // everyone done reading buf before it gets overwritten
    buf ^= 1;
  }

  const int crow = (lane >> 4) << 3;
  const int ccol = lane & 15;
#pragma unroll
  for (int i = 0; i < 2; ++i) {
#pragma unroll
    for (int j = 0; j < 4; ++j) {
#pragma unroll
      for (int r = 0; r < 8; ++r) {
        size_t mm = (size_t)(mBlk + wm * 32 + i * 16 + crow + r);
        int    nn = nBlk + wn * 64 + j * 16 + ccol;
        C[mm * N + nn] = acc[i][j][r] + bias[nn];
      }
    }
  }
}

// ---------------- LIF update (dendritic branches, soft reset) ----------------
// cur: (B, O*4) row-major, column j = o*4 + branch; one thread per (b, o).
__global__ void k_lif(const float* __restrict__ cur, float* __restrict__ d,
                      float* __restrict__ mem, _Float16* __restrict__ spk,
                      const float* __restrict__ alpha, const float* __restrict__ beta,
                      int B, int O) {
  int i = blockIdx.x * blockDim.x + threadIdx.x;
  if (i >= B * O) return;
  int o = i % O;
  float4 c4 = *(const float4*)(cur + (size_t)i * 4);
  float4 dd = *(const float4*)(d + (size_t)i * 4);
  float4 be = *(const float4*)(beta + (size_t)o * 4);
  dd.x = be.x * dd.x + (1.f - be.x) * c4.x;
  dd.y = be.y * dd.y + (1.f - be.y) * c4.y;
  dd.z = be.z * dd.z + (1.f - be.z) * c4.z;
  dd.w = be.w * dd.w + (1.f - be.w) * c4.w;
  float s = dd.x + dd.y + dd.z + dd.w;
  float al = alpha[o];
  float m = mem[i];
  float sp = (float)spk[i];
  m = al * m + (1.f - al) * s - 1.0f * sp;     // VTH = 1
  float nsp = (m - 1.0f > 0.f) ? 1.f : 0.f;    // act_fun(mem - VTH)
  mem[i] = m;
  *(float4*)(d + (size_t)i * 4) = dd;
  spk[i] = (_Float16)nsp;
}

__global__ void k_finalize(const float* __restrict__ acc, float* __restrict__ out) {
  int i = blockIdx.x * blockDim.x + threadIdx.x;
  if (i >= BB * NCLS) return;
  int b = i / NCLS, c = i % NCLS;
  out[i] = acc[(size_t)b * NCLSP + c] * (1.0f / TT);
}

// ---------------- host launcher ----------------

extern "C" void kernel_launch(void* const* d_in, const int* in_sizes, int n_in,
                              void* d_out, int out_size, void* d_ws, size_t ws_size,
                              hipStream_t stream) {
  const float* dvs   = (const float*)d_in[0];
  const float* W1    = (const float*)d_in[1];
  const float* b1    = (const float*)d_in[2];
  const float* taum1 = (const float*)d_in[3];
  const float* taun1 = (const float*)d_in[4];
  const float* mask1 = (const float*)d_in[5];
  const float* W2    = (const float*)d_in[6];
  const float* b2    = (const float*)d_in[7];
  const float* taum2 = (const float*)d_in[8];
  const float* taun2 = (const float*)d_in[9];
  const float* mask2 = (const float*)d_in[10];
  const float* W3    = (const float*)d_in[11];
  const float* b3    = (const float*)d_in[12];
  const float* taum3 = (const float*)d_in[13];
  const float* taun3 = (const float*)d_in[14];
  const float* mask3 = (const float*)d_in[15];
  const float* W4    = (const float*)d_in[16];
  const float* b4    = (const float*)d_in[17];
  const float* mem10 = (const float*)d_in[18];
  const float* mem20 = (const float*)d_in[19];
  const float* mem30 = (const float*)d_in[20];
  float* out = (float*)d_out;

  char* p = (char*)d_ws;
  auto alloc = [&](size_t bytes) -> char* {
    char* r = p;
    p += (bytes + 255) & ~(size_t)255;
    return r;
  };

  _Float16* Xf   = (_Float16*)alloc((size_t)TT * BB * IN_DIM * 2);
  _Float16* Wm1f = (_Float16*)alloc((size_t)NH1 * 4 * IN_DIM * 2);
  _Float16* Wm2f = (_Float16*)alloc((size_t)NH2 * 4 * NH1 * 2);
  _Float16* Wm3f = (_Float16*)alloc((size_t)NH3 * 4 * NH2 * 2);
  _Float16* W4f  = (_Float16*)alloc((size_t)NCLSP * NH3 * 2);
  float* b4p     = (float*)alloc(NCLSP * 4);
  float* cur1    = (float*)alloc((size_t)TT * BB * NH1 * 4 * 4);
  float* cur2    = (float*)alloc((size_t)BB * NH2 * 4 * 4);
  float* cur3    = (float*)alloc((size_t)BB * NH3 * 4 * 4);
  float* d1      = (float*)alloc((size_t)BB * NH1 * 4 * 4);
  float* d2      = (float*)alloc((size_t)BB * NH2 * 4 * 4);
  float* d3      = (float*)alloc((size_t)BB * NH3 * 4 * 4);
  float* mem1    = (float*)alloc((size_t)BB * NH1 * 4);
  float* mem2    = (float*)alloc((size_t)BB * NH2 * 4);
  float* mem3    = (float*)alloc((size_t)BB * NH3 * 4);
  _Float16* spk1 = (_Float16*)alloc((size_t)BB * NH1 * 2);
  _Float16* spk2 = (_Float16*)alloc((size_t)BB * NH2 * 2);
  _Float16* spk3 = (_Float16*)alloc((size_t)BB * NH3 * 2);
  float* al1     = (float*)alloc(NH1 * 4);
  float* be1     = (float*)alloc(NH1 * 4 * 4);
  float* al2     = (float*)alloc(NH2 * 4);
  float* be2     = (float*)alloc(NH2 * 4 * 4);
  float* al3     = (float*)alloc(NH3 * 4);
  float* be3     = (float*)alloc(NH3 * 4 * 4);
  float* accb    = (float*)alloc((size_t)BB * NCLSP * 4);

  const int TPB = 256;
  auto nb = [](size_t n) { return (unsigned)((n + 255) / 256); };

  // ---- prep ----
  k_prep_weight<<<nb((size_t)NH1 * 4 * IN_DIM), TPB, 0, stream>>>(W1, mask1, Wm1f, NH1 * 4, NH1 * 4, IN_DIM);
  k_prep_weight<<<nb((size_t)NH2 * 4 * NH1), TPB, 0, stream>>>(W2, mask2, Wm2f, NH2 * 4, NH2 * 4, NH1);
  k_prep_weight<<<nb((size_t)NH3 * 4 * NH2), TPB, 0, stream>>>(W3, mask3, Wm3f, NH3 * 4, NH3 * 4, NH2);
  k_prep_weight<<<nb((size_t)NCLSP * NH3), TPB, 0, stream>>>(W4, nullptr, W4f, NCLS, NCLSP, NH3);
  k_prep_x<<<nb((size_t)TT * BB * IN_DIM), TPB, 0, stream>>>(dvs, Xf);
  k_sigmoid<<<nb(NH1), TPB, 0, stream>>>(taum1, al1, NH1);
  k_sigmoid<<<nb(NH1 * 4), TPB, 0, stream>>>(taun1, be1, NH1 * 4);
  k_sigmoid<<<nb(NH2), TPB, 0, stream>>>(taum2, al2, NH2);
  k_sigmoid<<<nb(NH2 * 4), TPB, 0, stream>>>(taun2, be2, NH2 * 4);
  k_sigmoid<<<nb(NH3), TPB, 0, stream>>>(taum3, al3, NH3);
  k_sigmoid<<<nb(NH3 * 4), TPB, 0, stream>>>(taun3, be3, NH3 * 4);
  k_pad_bias<<<1, NCLSP, 0, stream>>>(b4, b4p, NCLS, NCLSP);
  k_copy_f32<<<nb(BB * NH1), TPB, 0, stream>>>(mem10, mem1, BB * NH1);
  k_copy_f32<<<nb(BB * NH2), TPB, 0, stream>>>(mem20, mem2, BB * NH2);
  k_copy_f32<<<nb(BB * NH3), TPB, 0, stream>>>(mem30, mem3, BB * NH3);
  k_zero_f32<<<nb(BB * NH1 * 4), TPB, 0, stream>>>(d1, BB * NH1 * 4);
  k_zero_f32<<<nb(BB * NH2 * 4), TPB, 0, stream>>>(d2, BB * NH2 * 4);
  k_zero_f32<<<nb(BB * NH3 * 4), TPB, 0, stream>>>(d3, BB * NH3 * 4);
  k_zero_f16<<<nb(BB * NH1), TPB, 0, stream>>>(spk1, BB * NH1);
  k_zero_f16<<<nb(BB * NH2), TPB, 0, stream>>>(spk2, BB * NH2);
  k_zero_f16<<<nb(BB * NH3), TPB, 0, stream>>>(spk3, BB * NH3);
  k_zero_f32<<<nb(BB * NCLSP), TPB, 0, stream>>>(accb, BB * NCLSP);

  // ---- time-batched layer-1 GEMM (FLOP-dominant): async-LDS staged WMMA ----
  {
    dim3 g(TT * BB / 128, (NH1 * 4) / 128);
    k_gemm_wmma_lds<<<g, 256, 0, stream>>>(Xf, Wm1f, b1, cur1, TT * BB, NH1 * 4, IN_DIM);
  }

  // ---- sequential timestep pipeline (L2-resident operands, direct-load WMMA) ----
  for (int t = 0; t < TT; ++t) {
    k_lif<<<nb(BB * NH1), TPB, 0, stream>>>(cur1 + (size_t)t * BB * NH1 * 4,
                                            d1, mem1, spk1, al1, be1, BB, NH1);
    {
      dim3 g(1, (NH2 * 4) / 128);
      k_gemm_wmma<<<g, 256, 0, stream>>>(spk1, Wm2f, b2, cur2, BB, NH2 * 4, NH1, 0);
    }
    k_lif<<<nb(BB * NH2), TPB, 0, stream>>>(cur2, d2, mem2, spk2, al2, be2, BB, NH2);
    {
      dim3 g(1, (NH3 * 4) / 128);
      k_gemm_wmma<<<g, 256, 0, stream>>>(spk2, Wm3f, b3, cur3, BB, NH3 * 4, NH2, 0);
    }
    k_lif<<<nb(BB * NH3), TPB, 0, stream>>>(cur3, d3, mem3, spk3, al3, be3, BB, NH3);
    {
      dim3 g(1, NCLSP / 128);
      k_gemm_wmma<<<g, 256, 0, stream>>>(spk3, W4f, b4p, accb, BB, NCLSP, NH3, 1);
    }
  }

  k_finalize<<<nb(BB * NCLS), TPB, 0, stream>>>(accb, out);
}
